// JointBiLSTM_7791070675414
// MI455X (gfx1250) — matrix operand order repaired and checked by reading, not compiled, not run on hardware
//
#include <hip/hip_runtime.h>

// ---------------------------------------------------------------------------
// JointBiLSTM for MI455X (gfx1250, wave32, WMMA).
// All GEMMs use v_wmma_f32_16x16x32_bf16 with fp32 accumulation. Weights are
// pre-transposed to [K x N] bf16; A fragments are two 128-bit loads, B fragments 16 strided loads.
// GEMM waves compute 32x64 tiles (8 accumulators, software-pipelined loads)
// with pointer-bump addressing. LSTM cell state stays in registers across all
// 128 timesteps; h is exchanged through 12KB of LDS per direction-workgroup.
// ---------------------------------------------------------------------------

typedef __bf16 bf16_t;
typedef __attribute__((ext_vector_type(16))) __bf16 v16bf;
typedef __attribute__((ext_vector_type(8)))  __bf16 v8bf;
typedef __attribute__((ext_vector_type(8)))  float  v8f;

constexpr int cE  = 384;          // LSTM hidden per direction
constexpr int cH  = 768;          // 2*cE
constexpr int cG  = 1536;         // 4*cE gate rows
constexpr int cB  = 8;
constexpr int cT  = 128;
constexpr int cBT = cB * cT;      // 1024
constexpr int cL  = 9;

// ------------------------------ WMMA helpers -------------------------------

__device__ __forceinline__ v8f wmma_bf16(v16bf a, v16bf b, v8f c) {
  return __builtin_amdgcn_wmma_f32_16x16x32_bf16(false, a, false, b,
                                                 (short)0, c, false, false);
}

__device__ __forceinline__ v16bf cat8(v8bf lo, v8bf hi) {
  return __builtin_shufflevector(lo, hi, 0, 1, 2, 3, 4, 5, 6, 7,
                                 8, 9, 10, 11, 12, 13, 14, 15);
}

// Generic fragment: two 16B loads at p and p+off2 (elements).
// A-frag (16x32 of row-major A): p = row base + k0 + (lane>>4)*8, off2 = 16.
// B-frag (row-major B[KxN]):     p = (k0+(lane>>4)*8)*ldb + col0 + (lane&15), strided.
__device__ __forceinline__ v16bf frag16(const bf16_t* p, int off2) {
  return cat8(*(const v8bf*)p, *(const v8bf*)(p + off2));
}

__device__ __forceinline__ v16bf fragB(const bf16_t* p, int ldb) {
  v16bf r;
#pragma unroll
  for (int s = 0; s < 8; ++s) {
    r[s] = p[(size_t)s * ldb];
    r[8 + s] = p[(size_t)(16 + s) * ldb];
  }
  return r;
}

// C/D fragment store: vgpr j -> m = row0 + j + 8*(lane>>4), lane&15 -> n.
__device__ __forceinline__ void store_tile(float* Cf, int ldc, int row0,
                                           int col0, int lane, v8f acc, float bias) {
  const int n = col0 + (lane & 15);
  const int mbase = row0 + ((lane >> 4) << 3);
#pragma unroll
  for (int j = 0; j < 8; ++j) Cf[(size_t)(mbase + j) * ldc + n] = acc[j] + bias;
}

// Software-pipelined 32x64 strip: acc[0..7] += A(32xK) @ B(Kx64 at col0).
// acc[q] = rows row0..+15, acc[4+q] = rows row0+16..+31, cols col0+16q.
__device__ __forceinline__ void gemm_strip(const bf16_t* __restrict__ A, int lda,
                                           const bf16_t* __restrict__ Bm, int ldb,
                                           int row0, int col0, int K, int lane,
                                           v8f acc[8]) {
  const bf16_t* pa0 = A + (size_t)(row0 + (lane & 15)) * lda + ((lane >> 4) << 3);
  const bf16_t* pa1 = pa0 + (size_t)16 * lda;
  const bf16_t* pb  = Bm + (size_t)((lane >> 4) << 3) * ldb + col0 + (lane & 15);
  v16bf a0 = frag16(pa0, 16);
  v16bf a1 = frag16(pa1, 16);
  v16bf b0 = fragB(pb, ldb);
  v16bf b1 = fragB(pb + 16, ldb);
  v16bf b2 = fragB(pb + 32, ldb);
  v16bf b3 = fragB(pb + 48, ldb);
  for (int k0 = 32; k0 < K; k0 += 32) {
    pa0 += 32;
    pa1 += 32;
    pb += (size_t)32 * ldb;
    v16bf an0 = frag16(pa0, 16);        // next-step loads issue before the
    v16bf an1 = frag16(pa1, 16);        // current WMMAs consume theirs
    v16bf c0 = fragB(pb, ldb);
    v16bf c1 = fragB(pb + 16, ldb);
    v16bf c2 = fragB(pb + 32, ldb);
    v16bf c3 = fragB(pb + 48, ldb);
    acc[0] = wmma_bf16(a0, b0, acc[0]);
    acc[1] = wmma_bf16(a0, b1, acc[1]);
    acc[2] = wmma_bf16(a0, b2, acc[2]);
    acc[3] = wmma_bf16(a0, b3, acc[3]);
    acc[4] = wmma_bf16(a1, b0, acc[4]);
    acc[5] = wmma_bf16(a1, b1, acc[5]);
    acc[6] = wmma_bf16(a1, b2, acc[6]);
    acc[7] = wmma_bf16(a1, b3, acc[7]);
    a0 = an0; a1 = an1; b0 = c0; b1 = c1; b2 = c2; b3 = c3;
  }
  acc[0] = wmma_bf16(a0, b0, acc[0]);
  acc[1] = wmma_bf16(a0, b1, acc[1]);
  acc[2] = wmma_bf16(a0, b2, acc[2]);
  acc[3] = wmma_bf16(a0, b3, acc[3]);
  acc[4] = wmma_bf16(a1, b0, acc[4]);
  acc[5] = wmma_bf16(a1, b1, acc[5]);
  acc[6] = wmma_bf16(a1, b2, acc[6]);
  acc[7] = wmma_bf16(a1, b3, acc[7]);
}

// ------------------------------ GEMM kernels -------------------------------

// C[M x N] = A[M x K](bf16) @ B[K x N](bf16, pre-transposed weight) + biases.
// 256 threads = 8 waves; each wave does a 32x64 strip.  M % 32 == 0.
__global__ __launch_bounds__(256)
void k_gemm_ab(const bf16_t* __restrict__ A, const bf16_t* __restrict__ Bm,
               const float* __restrict__ bias0, const float* __restrict__ bias1,
               float* __restrict__ Cf, int M, int N, int K) {
  const int lane = threadIdx.x & 31, wave = threadIdx.x >> 5;
  const int nj = N >> 6;
  const int job = blockIdx.x * 8 + wave;
  if (job >= (M >> 5) * nj) return;
  const int row0 = (job / nj) << 5, col0 = (job % nj) << 6;
  v8f acc[8] = {};
  gemm_strip(A, K, Bm, N, row0, col0, K, lane, acc);
#pragma unroll
  for (int q = 0; q < 4; ++q) {
    const int c0 = col0 + (q << 4);
    const int n = c0 + (lane & 15);
    const float bias = (bias0 ? bias0[n] : 0.f) + (bias1 ? bias1[n] : 0.f);
    store_tile(Cf, N, row0,      c0, lane, acc[q],     bias);
    store_tile(Cf, N, row0 + 16, c0, lane, acc[4 + q], bias);
  }
}

// Batched C[b] = A[b][M x K] @ B[b][K x N] (both bf16 row-major), fp32 out.
__global__ __launch_bounds__(256)
void k_gemm_ab_batched(const bf16_t* __restrict__ A, const bf16_t* __restrict__ Bm,
                       float* __restrict__ Cf, int M, int N, int K,
                       size_t sA, size_t sB, size_t sC) {
  const int bb = blockIdx.y;
  const bf16_t* Ab = A + (size_t)bb * sA;
  const bf16_t* Bb = Bm + (size_t)bb * sB;
  float* Cb = Cf + (size_t)bb * sC;
  const int lane = threadIdx.x & 31, wave = threadIdx.x >> 5;
  const int nj = N >> 6;
  const int job = blockIdx.x * 8 + wave;
  if (job >= (M >> 5) * nj) return;
  const int row0 = (job / nj) << 5, col0 = (job % nj) << 6;
  v8f acc[8] = {};
  gemm_strip(Ab, K, Bb, N, row0, col0, K, lane, acc);
#pragma unroll
  for (int q = 0; q < 4; ++q) {
    const int c0 = col0 + (q << 4);
    store_tile(Cb, N, row0,      c0, lane, acc[q],     0.f);
    store_tile(Cb, N, row0 + 16, c0, lane, acc[4 + q], 0.f);
  }
}

// --------------------------- LSTM recurrence -------------------------------
// grid.x = direction (0=fwd, 1=bwd). 256 threads = 8 waves.
// Each wave owns 3 e-tiles (24 total) and all 4 gates for them, so the cell
// state c lives in registers for all 128 steps (C layout: vgpr<->batch,
// lane<->e). h (16x384 bf16, rows 8..15 zero-padded) lives in LDS.
// Whht is the pre-transposed recurrent weight [cE x cG]; the four gate
// fragments are immediate offsets off one per-lane row base (one clause).
__global__ __launch_bounds__(256)
void k_lstm_recur(const bf16_t* __restrict__ Whht_all,  // [2][cE][cG]
                  const float* __restrict__ Xg_all,     // [2][cBT][cG]
                  bf16_t* __restrict__ hout_bf,         // [cBT][cH]
                  float* __restrict__ hout_f) {         // nullable [cBT][cH]
  const int dir = blockIdx.x;
  const bf16_t* Whht = Whht_all + (size_t)dir * cE * cG;
  const float*  Xg   = Xg_all + (size_t)dir * (size_t)cBT * cG;
  __shared__ __align__(16) bf16_t hshr[16 * cE];
  const int tid = threadIdx.x, lane = tid & 31, wave = tid >> 5;
  for (int i = tid; i < 16 * cE; i += 256) hshr[i] = (bf16_t)0.f;
  v8f cst[3] = {};
  __syncthreads();

  const bf16_t* pa0 = hshr + (size_t)(lane & 15) * cE + ((lane >> 4) << 3);
  const bf16_t* pw0 = Whht + (size_t)((lane >> 4) << 3) * cG + (lane & 15);

  for (int t = 0; t < cT; ++t) {
    const int trow = dir ? (cT - 1 - t) : t;
    v8f hnew[3] = {};
#pragma unroll 1
    for (int i = 0; i < 3; ++i) {
      const int e0 = (wave + (i << 3)) << 4;
      const bf16_t* pa = pa0;
      const bf16_t* pw = pw0 + e0;
      v8f acc[4] = {};
      v16bf a  = frag16(pa, 16);
      v16bf b0 = fragB(pw, cG);
      v16bf b1 = fragB(pw + cE, cG);
      v16bf b2 = fragB(pw + 2 * cE, cG);
      v16bf b3 = fragB(pw + 3 * cE, cG);
      for (int k0 = 32; k0 < cE; k0 += 32) {
        pa += 32;
        pw += (size_t)32 * cG;
        v16bf an = frag16(pa, 16);
        v16bf c0 = fragB(pw, cG);
        v16bf c1 = fragB(pw + cE, cG);
        v16bf c2 = fragB(pw + 2 * cE, cG);
        v16bf c3 = fragB(pw + 3 * cE, cG);
        acc[0] = wmma_bf16(a, b0, acc[0]);
        acc[1] = wmma_bf16(a, b1, acc[1]);
        acc[2] = wmma_bf16(a, b2, acc[2]);
        acc[3] = wmma_bf16(a, b3, acc[3]);
        a = an; b0 = c0; b1 = c1; b2 = c2; b3 = c3;
      }
      acc[0] = wmma_bf16(a, b0, acc[0]);
      acc[1] = wmma_bf16(a, b1, acc[1]);
      acc[2] = wmma_bf16(a, b2, acc[2]);
      acc[3] = wmma_bf16(a, b3, acc[3]);

      if (lane < 16) {                   // lanes 0..15 hold batches 0..7
        const int e = e0 + lane;
#pragma unroll
        for (int j = 0; j < 8; ++j) {    // j = batch
          const float* xg = Xg + (size_t)(j * cT + trow) * cG;
          float iv = acc[0][j] + xg[e];
          float fv = acc[1][j] + xg[cE + e];
          float gv = acc[2][j] + xg[2 * cE + e];
          float ov = acc[3][j] + xg[3 * cE + e];
          iv = 1.f / (1.f + __expf(-iv));
          fv = 1.f / (1.f + __expf(-fv));
          gv = tanhf(gv);
          ov = 1.f / (1.f + __expf(-ov));
          const float cc = fv * cst[i][j] + iv * gv;
          cst[i][j] = cc;
          hnew[i][j] = ov * tanhf(cc);
        }
      }
    }
    __syncthreads();                     // all hshr reads of step t done
    if (lane < 16) {
      for (int i = 0; i < 3; ++i) {
        const int e = ((wave + (i << 3)) << 4) + lane;
#pragma unroll
        for (int j = 0; j < 8; ++j) {
          const float h = hnew[i][j];
          hshr[j * cE + e] = (bf16_t)h;
          const size_t o = (size_t)(j * cT + trow) * cH + (size_t)dir * cE + e;
          hout_bf[o] = (bf16_t)h;
          if (hout_f) hout_f[o] = h;
        }
      }
    }
    __syncthreads();
  }
}

// --------------------------- small utility kernels -------------------------

__global__ void k_f2bf(const float* __restrict__ s, bf16_t* __restrict__ d, size_t n) {
  size_t i = (size_t)blockIdx.x * blockDim.x + threadIdx.x;
  if (i < n) d[i] = (bf16_t)s[i];
}

// Wt[k][n] = (bf16) W[n][k]  (tiled 32x32 transpose through LDS).
__global__ __launch_bounds__(256)
void k_f2bf_T(const float* __restrict__ W, bf16_t* __restrict__ Wt, int N, int K) {
  __shared__ float tile[32][33];
  const int tn0 = blockIdx.x * 32, tk0 = blockIdx.y * 32;
  const int tx = threadIdx.x & 31, ty = threadIdx.x >> 5;  // 32 x 8
#pragma unroll
  for (int r = 0; r < 4; ++r) {
    const int n = tn0 + ty + 8 * r, k = tk0 + tx;
    tile[ty + 8 * r][tx] = (n < N && k < K) ? W[(size_t)n * K + k] : 0.f;
  }
  __syncthreads();
#pragma unroll
  for (int r = 0; r < 4; ++r) {
    const int k = tk0 + ty + 8 * r, n = tn0 + tx;
    if (k < K && n < N) Wt[(size_t)k * N + n] = (bf16_t)tile[tx][ty + 8 * r];
  }
}

__global__ void k_zero(float* p, int n) {
  int i = blockIdx.x * blockDim.x + threadIdx.x;
  if (i < n) p[i] = 0.f;
}

// hp[32 x cH] bf16: rows 0..7 = hs_bf[b, qid[b], :], rows 8..31 zero.
__global__ void k_gather_hp(const bf16_t* __restrict__ hs_bf,
                            const int* __restrict__ qid, bf16_t* __restrict__ hp) {
  int i = blockIdx.x * blockDim.x + threadIdx.x;
  if (i >= 32 * cH) return;
  int r = i / cH, h = i % cH;
  bf16_t v = (bf16_t)0.f;
  if (r < cB) v = hs_bf[((size_t)r * cT + qid[r]) * cH + h];
  hp[i] = v;
}

// scores[b,t,s] = v . tanh(shared[b,s,:] + p[b,:] + tt[b,t,:]); softmax over s.
// block = (b,t); 128 threads, thread = s.  float4-vectorized inner loop.
__global__ __launch_bounds__(128)
void k_attn(const float* __restrict__ shared_f, const float* __restrict__ tterm_f,
            const float* __restrict__ pterm_f, const float* __restrict__ vvec,
            bf16_t* __restrict__ a_bf, float* __restrict__ at_pool) {
  const int bt = blockIdx.x, b = bt / cT;
  const int s = threadIdx.x;
  const float4* sh = (const float4*)(shared_f + (size_t)(b * cT + s) * cH);
  const float4* tt = (const float4*)(tterm_f + (size_t)bt * cH);
  const float4* pp = (const float4*)(pterm_f + (size_t)b * cH);
  const float4* vv = (const float4*)vvec;
  float acc = 0.f;
  for (int h = 0; h < cH / 4; ++h) {
    const float4 a = sh[h], t4 = tt[h], p4 = pp[h], w = vv[h];
    acc += w.x * tanhf(a.x + t4.x + p4.x);
    acc += w.y * tanhf(a.y + t4.y + p4.y);
    acc += w.z * tanhf(a.z + t4.z + p4.z);
    acc += w.w * tanhf(a.w + t4.w + p4.w);
  }
  __shared__ float buf[cT];
  __shared__ float smax, ssum;
  buf[s] = acc; __syncthreads();
  for (int off = 64; off > 0; off >>= 1) {
    if (s < off) buf[s] = fmaxf(buf[s], buf[s + off]);
    __syncthreads();
  }
  if (s == 0) smax = buf[0];
  __syncthreads();
  const float ex = __expf(acc - smax);
  buf[s] = ex; __syncthreads();
  for (int off = 64; off > 0; off >>= 1) {
    if (s < off) buf[s] += buf[s + off];
    __syncthreads();
  }
  if (s == 0) ssum = buf[0];
  __syncthreads();
  const float p = ex / ssum;
  a_bf[(size_t)bt * cT + s] = (bf16_t)p;
  atomicAdd(&at_pool[b * cT + s], p * (1.f / cT));
}

// norm = at_pool / sum_s at_pool, nan -> 0.
__global__ __launch_bounds__(128)
void k_norm(const float* __restrict__ at_pool, float* __restrict__ out_at) {
  const int b = blockIdx.x, s = threadIdx.x;
  __shared__ float buf[cT];
  __shared__ float ssum;
  const float v = at_pool[b * cT + s];
  buf[s] = v; __syncthreads();
  for (int off = 64; off > 0; off >>= 1) {
    if (s < off) buf[s] += buf[s + off];
    __syncthreads();
  }
  if (s == 0) ssum = buf[0];
  __syncthreads();
  float r = v / ssum;
  if (r != r) r = 0.f;
  out_at[b * cT + s] = r;
}

// logits[bt][l] = hs[bt].W_clf[l][0:H] + c[bt].W_clf[l][H:2H]; 9 waves/block.
__global__ __launch_bounds__(288)
void k_logits(const float* __restrict__ hs_f, const float* __restrict__ c_f,
              const float* __restrict__ Wclf, float* __restrict__ logits) {
  const int bt = blockIdx.x;
  const int wave = threadIdx.x >> 5, lane = threadIdx.x & 31;
  if (wave >= cL) return;
  const float* u0 = hs_f + (size_t)bt * cH;
  const float* u1 = c_f + (size_t)bt * cH;
  const float* w = Wclf + (size_t)wave * (2 * cH);
  float acc = 0.f;
  for (int h = lane; h < cH; h += 32) acc += u0[h] * w[h] + u1[h] * w[cH + h];
  for (int off = 16; off > 0; off >>= 1) acc += __shfl_down(acc, off, 32);
  if (lane == 0) logits[(size_t)bt * cL + wave] = acc;
}

// Viterbi per batch; writes vit_logits region as floats (+10 / -1).
__global__ __launch_bounds__(32)
void k_viterbi(const float* __restrict__ logits, const float* __restrict__ trans,
               const float* __restrict__ cstart, const float* __restrict__ cend,
               int* __restrict__ ptrs, float* __restrict__ out_vit) {
  const int b = blockIdx.x, tid = threadIdx.x;
  __shared__ float sc[cL], nsc[cL];
  __shared__ int tagbuf[cT];
  if (tid < cL) sc[tid] = cstart[tid] + logits[(size_t)(b * cT) * cL + tid];
  __syncthreads();
  for (int t = 1; t < cT; ++t) {
    if (tid < cL) {
      float best = -1e30f; int bi = 0;
      for (int i = 0; i < cL; ++i) {
        const float cand = sc[i] + trans[i * cL + tid];
        if (cand > best) { best = cand; bi = i; }
      }
      nsc[tid] = best + logits[(size_t)(b * cT + t) * cL + tid];
      ptrs[((size_t)b * cT + t) * cL + tid] = bi;
    }
    __syncthreads();
    if (tid < cL) sc[tid] = nsc[tid];
    __syncthreads();
  }
  if (tid == 0) {
    float best = -1e30f; int bi = 0;
    for (int i = 0; i < cL; ++i) {
      const float f = sc[i] + cend[i];
      if (f > best) { best = f; bi = i; }
    }
    tagbuf[cT - 1] = bi;
    for (int t = cT - 1; t > 0; --t) {
      bi = ptrs[((size_t)b * cT + t) * cL + bi];
      tagbuf[t - 1] = bi;
    }
  }
  __syncthreads();
  for (int i = tid; i < cT * cL; i += 32) {
    const int t = i / cL, l = i % cL;
    out_vit[((size_t)b * cT + t) * cL + l] = (tagbuf[t] == l) ? 10.f : -1.f;
  }
}

// ------------------------------- host side ---------------------------------

extern "C" void kernel_launch(void* const* d_in, const int* in_sizes, int n_in,
                              void* d_out, int out_size, void* d_ws, size_t ws_size,
                              hipStream_t stream) {
  (void)in_sizes; (void)n_in; (void)out_size; (void)ws_size;
  const float* x_emb   = (const float*)d_in[0];
  const int*   qids    = (const int*)  d_in[1];
  const float* w_ih_l0 = (const float*)d_in[2];   // [2][cG][cE]
  const float* w_ih_r  = (const float*)d_in[3];   // [2][2][cG][cH]
  const float* w_hh    = (const float*)d_in[4];   // [3][2][cG][cE]
  const float* b_ih    = (const float*)d_in[5];   // [3][2][cG]
  const float* b_hh    = (const float*)d_in[6];
  const float* W_H     = (const float*)d_in[7];
  const float* W_p     = (const float*)d_in[8];
  const float* W_h     = (const float*)d_in[9];
  const float* vvec    = (const float*)d_in[10];
  const float* W_clf   = (const float*)d_in[11];
  const float* trans   = (const float*)d_in[12];
  const float* cstart  = (const float*)d_in[13];
  const float* cend    = (const float*)d_in[14];

  float* out     = (float*)d_out;                 // vit | hidden | a_t
  float* out_vit = out;
  float* out_hid = out + (size_t)cB * cT * cL;
  float* out_at  = out_hid + (size_t)cBT * cH;

  // ---- workspace carve (~48 MB) ----
  char* cur = (char*)d_ws;
  auto carve = [&](size_t bytes) -> void* {
    void* p = (void*)cur;
    cur += (bytes + 255) & ~(size_t)255;
    return p;
  };
  bf16_t* wih0_t  = (bf16_t*)carve(sizeof(bf16_t) * 2ull * cE * cG);   // [dir][cE][cG]
  bf16_t* wihR_t  = (bf16_t*)carve(sizeof(bf16_t) * 4ull * cH * cG);   // [l-1,dir][cH][cG]
  bf16_t* whh_t   = (bf16_t*)carve(sizeof(bf16_t) * 6ull * cE * cG);   // [l,dir][cE][cG]
  bf16_t* WH_t    = (bf16_t*)carve(sizeof(bf16_t) * (size_t)cH * cH);
  bf16_t* Wp_t    = (bf16_t*)carve(sizeof(bf16_t) * (size_t)cH * cH);
  bf16_t* Wh_t    = (bf16_t*)carve(sizeof(bf16_t) * (size_t)cH * cH);
  bf16_t* xin0_bf = (bf16_t*)carve(sizeof(bf16_t) * (size_t)cBT * cE);
  bf16_t* hbufA   = (bf16_t*)carve(sizeof(bf16_t) * (size_t)cBT * cH);
  bf16_t* hbufB   = (bf16_t*)carve(sizeof(bf16_t) * (size_t)cBT * cH);
  float*  Xg      = (float*) carve(sizeof(float) * 2ull * cBT * cG);
  float*  shr_f   = (float*) carve(sizeof(float) * (size_t)cBT * cH);
  float*  tterm_f = (float*) carve(sizeof(float) * (size_t)cBT * cH);
  bf16_t* hp_bf   = (bf16_t*)carve(sizeof(bf16_t) * 32ull * cH);
  float*  pterm_f = (float*) carve(sizeof(float) * 32ull * cH);
  bf16_t* a_bf    = (bf16_t*)carve(sizeof(bf16_t) * (size_t)cB * cT * cT);
  float*  at_pool = (float*) carve(sizeof(float) * (size_t)cB * cT);
  float*  c_f     = (float*) carve(sizeof(float) * (size_t)cBT * cH);
  float*  logits_f= (float*) carve(sizeof(float) * (size_t)cBT * cL);
  int*    ptrs    = (int*)   carve(sizeof(int)   * (size_t)cB * cT * cL);

  auto cvtT = [&](const float* W, bf16_t* Wt, int N, int K) {
    k_f2bf_T<<<dim3((N + 31) / 32, (K + 31) / 32), 256, 0, stream>>>(W, Wt, N, K);
  };
  // ---- weight transposes + input conversion to bf16 ----
  for (int d = 0; d < 2; ++d)
    cvtT(w_ih_l0 + (size_t)d * cG * cE, wih0_t + (size_t)d * cE * cG, cG, cE);
  for (int i = 0; i < 4; ++i)
    cvtT(w_ih_r + (size_t)i * cG * cH, wihR_t + (size_t)i * cH * cG, cG, cH);
  for (int i = 0; i < 6; ++i)
    cvtT(w_hh + (size_t)i * cG * cE, whh_t + (size_t)i * cE * cG, cG, cE);
  cvtT(W_H, WH_t, cH, cH);
  cvtT(W_p, Wp_t, cH, cH);
  cvtT(W_h, Wh_t, cH, cH);
  k_f2bf<<<(cBT * cE + 255) / 256, 256, 0, stream>>>(x_emb, xin0_bf, (size_t)cBT * cE);

  const int gXg = (32 * (cG / 64) + 7) / 8;   // (M/32)*(N/64) jobs / 8 waves
  // ---- layer 0 ----
  for (int d = 0; d < 2; ++d)
    k_gemm_ab<<<gXg, 256, 0, stream>>>(xin0_bf, wih0_t + (size_t)d * cE * cG,
        b_ih + (size_t)d * cG, b_hh + (size_t)d * cG,
        Xg + (size_t)d * cBT * cG, cBT, cG, cE);
  k_lstm_recur<<<2, 256, 0, stream>>>(whh_t, Xg, hbufA, nullptr);
  // ---- layer 1 ----
  for (int d = 0; d < 2; ++d)
    k_gemm_ab<<<gXg, 256, 0, stream>>>(hbufA, wihR_t + (size_t)d * cH * cG,
        b_ih + (size_t)(2 + d) * cG, b_hh + (size_t)(2 + d) * cG,
        Xg + (size_t)d * cBT * cG, cBT, cG, cH);
  k_lstm_recur<<<2, 256, 0, stream>>>(whh_t + 2ull * cE * cG, Xg, hbufB, nullptr);
  // ---- layer 2 (also writes fp32 hidden_states output) ----
  for (int d = 0; d < 2; ++d)
    k_gemm_ab<<<gXg, 256, 0, stream>>>(hbufB, wihR_t + (size_t)(2 + d) * cH * cG,
        b_ih + (size_t)(4 + d) * cG, b_hh + (size_t)(4 + d) * cG,
        Xg + (size_t)d * cBT * cG, cBT, cG, cH);
  k_lstm_recur<<<2, 256, 0, stream>>>(whh_t + 4ull * cE * cG, Xg, hbufA, out_hid);

  // ---- attention ----
  const int gHH = (32 * (cH / 64) + 7) / 8;    // 1024x768x768 GEMM
  k_gemm_ab<<<gHH, 256, 0, stream>>>(hbufA, WH_t, nullptr, nullptr, shr_f,   cBT, cH, cH);
  k_gemm_ab<<<gHH, 256, 0, stream>>>(hbufA, Wh_t, nullptr, nullptr, tterm_f, cBT, cH, cH);
  k_gather_hp<<<(32 * cH + 255) / 256, 256, 0, stream>>>(hbufA, qids, hp_bf);
  k_gemm_ab<<<2, 256, 0, stream>>>(hp_bf, Wp_t, nullptr, nullptr, pterm_f, 32, cH, cH);
  k_zero<<<(cB * cT + 255) / 256, 256, 0, stream>>>(at_pool, cB * cT);
  k_attn<<<cBT, cT, 0, stream>>>(shr_f, tterm_f, pterm_f, vvec, a_bf, at_pool);
  k_gemm_ab_batched<<<dim3(6, cB), 256, 0, stream>>>(a_bf, hbufA, c_f,
      cT, cH, cT, (size_t)cT * cT, (size_t)cT * cH, (size_t)cT * cH);
  k_norm<<<cB, cT, 0, stream>>>(at_pool, out_at);

  // ---- classifier + CRF ----
  k_logits<<<cBT, 288, 0, stream>>>(out_hid, c_f, W_clf, logits_f);
  k_viterbi<<<cB, 32, 0, stream>>>(logits_f, trans, cstart, cend, ptrs, out_vit);
}
